// ResidualStack_32959579029511
// MI455X (gfx1250) — compile-verified, matrix operand order checked by reading, not run on hardware
//
#include <hip/hip_runtime.h>
#include <hip/hip_bf16.h>
#include <math.h>

// ---------------- problem constants ----------------
#define N_LAYERS 10
#define N_STACKS 2
#define N_BLOCKS 20
#define RES_CH   128
#define BATCH    8
#define T_IN     8192
#define SKIP     6146

#define NT       32           // time-tile per workgroup
#define LDSP     136          // padded row length (bf16 elems), 272B stride, 16B-aligned rows

typedef __attribute__((ext_vector_type(8)))  float  v8f;
typedef __attribute__((ext_vector_type(8)))  __bf16 v8bf;
typedef __attribute__((ext_vector_type(16))) __bf16 v16bf;

union Frag { v16bf v; v8bf h[2]; };

static __device__ __forceinline__ unsigned short f2bf(float f) {
    union { float f; unsigned u; } x; x.f = f;
    unsigned r = x.u + 0x7fffu + ((x.u >> 16) & 1u);   // round-to-nearest-even
    return (unsigned short)(r >> 16);
}
static __device__ __forceinline__ float bf2f(unsigned short b) {
    union { unsigned u; float f; } x; x.u = ((unsigned)b) << 16; return x.f;
}

// ---------------- weight prep: fp32 -> bf16 hi/lo, de-interleave taps ----------------
// layout per block (each 128x128 row-major bf16): [Wd0h, Wd0l, Wd1h, Wd1l, W1h, W1l]
__global__ void wn_prep_weights(const float* __restrict__ Wd,
                                const float* __restrict__ W1,
                                unsigned short* __restrict__ wout) {
    int i = blockIdx.x * blockDim.x + threadIdx.x;
    if (i >= N_BLOCKS * RES_CH * RES_CH) return;
    int blk = i / (RES_CH * RES_CH);
    int rem = i - blk * (RES_CH * RES_CH);          // o*128 + ic
    float w0 = Wd[(size_t)i * 2 + 0];
    float w1 = Wd[(size_t)i * 2 + 1];
    float wc = W1[(size_t)i];
    unsigned short* base = wout + (size_t)blk * 6 * (RES_CH * RES_CH);
    unsigned short h;
    h = f2bf(w0); base[0 * 16384 + rem] = h; base[1 * 16384 + rem] = f2bf(w0 - bf2f(h));
    h = f2bf(w1); base[2 * 16384 + rem] = h; base[3 * 16384 + rem] = f2bf(w1 - bf2f(h));
    h = f2bf(wc); base[4 * 16384 + rem] = h; base[5 * 16384 + rem] = f2bf(wc - bf2f(h));
}

// ---------------- split-precision bf16 WMMA GEMM panel ----------------
// acc[sub] += W[16w:16w+16, :128] @ B[:128, 16*sub : 16*sub+16]  (3-term hi/lo split)
static __device__ __forceinline__ void gemm_panel(
        v8f acc[2],
        const unsigned short* __restrict__ Wh,     // 128x128 row-major bf16 (hi)
        const unsigned short* __restrict__ Wl,     // lo
        unsigned short (*Bh)[LDSP],                // LDS [t][ic] bf16 (hi)
        unsigned short (*Bl)[LDSP],                // lo
        int w, int nn, int half) {
#pragma unroll
    for (int kc = 0; kc < 4; ++kc) {
        int kb = 32 * kc + 8 * half;
        size_t ao = (size_t)(16 * w + nn) * RES_CH + kb;
        Frag ah, al;
        ah.h[0] = *(const v8bf*)(Wh + ao);
        ah.h[1] = *(const v8bf*)(Wh + ao + 16);
        al.h[0] = *(const v8bf*)(Wl + ao);
        al.h[1] = *(const v8bf*)(Wl + ao + 16);
#pragma unroll
        for (int sub = 0; sub < 2; ++sub) {
            int tt = 16 * sub + nn;
            Frag bh, bl;
            bh.h[0] = *(const v8bf*)(&Bh[tt][kb]);
            bh.h[1] = *(const v8bf*)(&Bh[tt][kb + 16]);
            bl.h[0] = *(const v8bf*)(&Bl[tt][kb]);
            bl.h[1] = *(const v8bf*)(&Bl[tt][kb + 16]);
            acc[sub] = __builtin_amdgcn_wmma_f32_16x16x32_bf16(
                false, ah.v, false, bh.v, (short)0, acc[sub], false, false);
            acc[sub] = __builtin_amdgcn_wmma_f32_16x16x32_bf16(
                false, ah.v, false, bl.v, (short)0, acc[sub], false, false);
            acc[sub] = __builtin_amdgcn_wmma_f32_16x16x32_bf16(
                false, al.v, false, bh.v, (short)0, acc[sub], false, false);
        }
    }
}

// ---------------- one residual block ----------------
// grid: ((T_out+NT-1)/NT, BATCH), block: 256 threads (8 wave32)
__global__ __launch_bounds__(256)
void wn_resblock(const float* __restrict__ cur,    // [B,128,T_cur]
                 float* __restrict__ nxt,          // [B,128,T_out]
                 const unsigned short* __restrict__ wblk, // 6*16384 bf16
                 const float* __restrict__ bd,     // [128]
                 const float* __restrict__ b1,     // [128]
                 float* __restrict__ skip,         // [B,128,SKIP] for this block
                 int T_cur, int T_out, int d) {
    __shared__ __align__(16) unsigned short sXLh[NT][LDSP];
    __shared__ __align__(16) unsigned short sXLl[NT][LDSP];
    __shared__ __align__(16) unsigned short sXRh[NT][LDSP];
    __shared__ __align__(16) unsigned short sXRl[NT][LDSP];
    __shared__ __align__(16) unsigned short sGh[NT][LDSP];
    __shared__ __align__(16) unsigned short sGl[NT][LDSP];

    const int tid  = threadIdx.x;
    const int b    = blockIdx.y;
    const int t0   = blockIdx.x * NT;
    const int w    = tid >> 5;
    const int lane = tid & 31;
    const int nn   = lane & 15;
    const int half = lane >> 4;

    // ---- stage input tiles (fp32 -> bf16 hi/lo), clamped at the tail ----
    for (int idx = tid; idx < NT * RES_CH; idx += 256) {
        int t  = idx >> 7;
        int ic = idx & 127;
        int tl = t0 + t;     if (tl > T_cur - 1) tl = T_cur - 1;
        int tr = t0 + t + d; if (tr > T_cur - 1) tr = T_cur - 1;
        const float* row = cur + (size_t)(b * RES_CH + ic) * T_cur;
        float vl = row[tl];
        float vr = row[tr];
        unsigned short hl = f2bf(vl);
        sXLh[t][ic] = hl; sXLl[t][ic] = f2bf(vl - bf2f(hl));
        unsigned short hr = f2bf(vr);
        sXRh[t][ic] = hr; sXRl[t][ic] = f2bf(vr - bf2f(hr));
    }
    __syncthreads();

    // ---- GEMM 1: g = Wd0 @ xl + Wd1 @ xr + bd ----
    v8f acc[2];
    const int mbase = 16 * w + 8 * half;
#pragma unroll
    for (int r = 0; r < 8; ++r) {
        float bv = bd[mbase + r];
        acc[0][r] = bv; acc[1][r] = bv;
    }
    gemm_panel(acc, wblk + 0 * 16384, wblk + 1 * 16384, sXLh, sXLl, w, nn, half);
    gemm_panel(acc, wblk + 2 * 16384, wblk + 3 * 16384, sXRh, sXRl, w, nn, half);

    // ---- gate = tanh(g) * sigmoid(g); stage to LDS as B of GEMM 2 ----
#pragma unroll
    for (int sub = 0; sub < 2; ++sub) {
        int tt = 16 * sub + nn;
#pragma unroll
        for (int r = 0; r < 8; ++r) {
            float g = acc[sub][r];
            float val = tanhf(g) * (1.0f / (1.0f + __expf(-g)));
            int m = mbase + r;
            unsigned short h = f2bf(val);
            sGh[tt][m] = h;
            sGl[tt][m] = f2bf(val - bf2f(h));
        }
    }
    __syncthreads();

    // ---- GEMM 2: c = W1 @ gate + b1 ----
#pragma unroll
    for (int r = 0; r < 8; ++r) {
        float bv = b1[mbase + r];
        acc[0][r] = bv; acc[1][r] = bv;
    }
    gemm_panel(acc, wblk + 4 * 16384, wblk + 5 * 16384, sGh, sGl, w, nn, half);

    // ---- residual + skip stores (coalesced along t) ----
    const int skip_off = T_out - SKIP;
#pragma unroll
    for (int sub = 0; sub < 2; ++sub) {
        int t = t0 + 16 * sub + nn;
        if (t < T_out) {
#pragma unroll
            for (int r = 0; r < 8; ++r) {
                int m = mbase + r;
                float c = acc[sub][r];
                float res = c + cur[(size_t)(b * RES_CH + m) * T_cur + t + d];
                nxt[(size_t)(b * RES_CH + m) * T_out + t] = res;
                int s = t - skip_off;
                if (s >= 0)
                    skip[(size_t)(b * RES_CH + m) * SKIP + s] = c;
            }
        }
    }
}

// ---------------- host-side launch ----------------
extern "C" void kernel_launch(void* const* d_in, const int* in_sizes, int n_in,
                              void* d_out, int out_size, void* d_ws, size_t ws_size,
                              hipStream_t stream) {
    const float* x  = (const float*)d_in[0];
    const float* Wd = (const float*)d_in[1];
    const float* bd = (const float*)d_in[2];
    const float* W1 = (const float*)d_in[3];
    const float* b1 = (const float*)d_in[4];
    float* out = (float*)d_out;

    // workspace: ping-pong activations + bf16 weights
    float* bufA = (float*)d_ws;
    float* bufB = bufA + (size_t)BATCH * RES_CH * T_IN;
    unsigned short* wbf = (unsigned short*)(bufB + (size_t)BATCH * RES_CH * T_IN);

    // split weights to bf16 hi/lo once (cheap; L2-resident afterwards)
    {
        int n = N_BLOCKS * RES_CH * RES_CH;
        wn_prep_weights<<<(n + 255) / 256, 256, 0, stream>>>(Wd, W1, wbf);
    }

    static const int DIL[N_BLOCKS] = {1, 2, 4, 8, 16, 32, 64, 128, 256, 512,
                                      1, 2, 4, 8, 16, 32, 64, 128, 256, 512};

    const float* cur = x;
    int T_cur = T_IN;
    for (int blk = 0; blk < N_BLOCKS; ++blk) {
        int d = DIL[blk];
        int T_out = T_cur - d;
        float* nxt = (blk & 1) ? bufB : bufA;
        dim3 grid((T_out + NT - 1) / NT, BATCH);
        wn_resblock<<<grid, 256, 0, stream>>>(
            cur, nxt,
            wbf + (size_t)blk * 6 * (RES_CH * RES_CH),
            bd + (size_t)blk * RES_CH,
            b1 + (size_t)blk * RES_CH,
            out + (size_t)blk * BATCH * RES_CH * SKIP,
            T_cur, T_out, d);
        cur = nxt;
        T_cur = T_out;
    }
}
